// SetEncoderPointNetSp_36636071035052
// MI455X (gfx1250) — matrix operand order repaired and checked by reading, not compiled, not run on hardware
//
#include <hip/hip_runtime.h>

// Problem constants (match the reference).
#define E_EDGES 1048576
#define C_IN    128
#define C_MID   256
#define C_OUT   128
#define NVERT   50000

typedef __attribute__((ext_vector_type(16))) __bf16        v16bf;
typedef __attribute__((ext_vector_type(4)))  __bf16        bf16x4;
typedef __attribute__((ext_vector_type(8)))  float         v8f;
typedef __attribute__((ext_vector_type(4)))  float         f32x4;
typedef __attribute__((ext_vector_type(4)))  unsigned int  u32x4;

union BF16Frag {
    unsigned short u[16];
    u32x4          q[2];
    bf16x4         b[4];
    v16bf          v;
};

// f32 -> bf16 via the native convert (backend emits v_cvt_(pk_)bf16_f32).
__device__ __forceinline__ unsigned short f2bf(float f) {
    return __builtin_bit_cast(unsigned short, (__bf16)f);
}

// Monotone (order-preserving) f32 -> u32 key so segment-max can use
// hardware global_atomic_max_u32. key(f) strictly increasing in f.
__device__ __forceinline__ unsigned fkey(float f) {
    unsigned u = __builtin_bit_cast(unsigned, f);
    return (u & 0x80000000u) ? ~u : (u | 0x80000000u);
}

// Inverse; key==0 (init sentinel) decodes to NaN bits -> forced to 0.0f,
// which exactly matches the reference's where(isfinite, z, 0).
__device__ __forceinline__ float funkey(unsigned k) {
    unsigned u = (k & 0x80000000u) ? (k & 0x7FFFFFFFu) : ~k;
    if ((u & 0x7F800000u) == 0x7F800000u) return 0.0f;  // inf/NaN/sentinel
    return __builtin_bit_cast(float, u);
}

__device__ __forceinline__ v8f wmma_bf16(v16bf a, v16bf b, v8f c) {
    return __builtin_amdgcn_wmma_f32_16x16x32_bf16(
        /*neg_a=*/false, a, /*neg_b=*/false, b,
        /*c_mod=*/(short)0, c, /*reuse_a=*/false, /*reuse_b=*/false);
}

// --- A fragment (16 rows x 32 K, bf16) from an f32 row-major source row. ---
// ISA layout (16-bit A 16x32): lane = h*16+ln holds row M=ln; VGPR 0-3 hold
// K = k0+8h .. k0+8h+7, VGPR 4-7 hold K = k0+16+8h .. +7 (two 8-K chunks).
__device__ __forceinline__ v16bf a_frag_from_f32(const float* rp, int k0, int h) {
    const f32x4* p0 = (const f32x4*)(rp + k0 + 8 * h);
    const f32x4* p1 = (const f32x4*)(rp + k0 + 16 + 8 * h);
    BF16Frag r;
    r.b[0] = __builtin_convertvector(p0[0], bf16x4);
    r.b[1] = __builtin_convertvector(p0[1], bf16x4);
    r.b[2] = __builtin_convertvector(p1[0], bf16x4);
    r.b[3] = __builtin_convertvector(p1[1], bf16x4);
    return r.v;
}

// A fragment from the segment-max key buffer (decode key -> f32 -> bf16).
__device__ __forceinline__ v16bf a_frag_from_keys(const unsigned* rp, int k0, int h) {
    const u32x4* p0 = (const u32x4*)(rp + k0 + 8 * h);
    const u32x4* p1 = (const u32x4*)(rp + k0 + 16 + 8 * h);
    u32x4 c[4] = {p0[0], p0[1], p1[0], p1[1]};
    BF16Frag r;
#pragma unroll
    for (int j = 0; j < 4; j++) {
        f32x4 d;
#pragma unroll
        for (int i = 0; i < 4; i++) d[i] = funkey(c[j][i]);
        r.b[j] = __builtin_convertvector(d, bf16x4);
    }
    return r.v;
}

// --- B fragment (32 K x 16 cols, bf16) from LDS, weights stored [col][K]. ---
// ISA layout (bf16 B): lane half h holds K = k0+8h..+7 then K = k0+16+8h..+7
// (two 16B chunks per lane, same K order as the A fragment).
__device__ __forceinline__ v16bf b_frag_from_lds(const unsigned short* sw, int col,
                                                 int stride, int k0, int h) {
    const u32x4* p0 = (const u32x4*)(sw + col * stride + k0 + 8 * h);
    const u32x4* p1 = (const u32x4*)(sw + col * stride + k0 + 16 + 8 * h);
    BF16Frag r;
    r.q[0] = p0[0];
    r.q[1] = p1[0];
    return r.v;
}

// ---------------------------------------------------------------------------
// Kernel 0: zero the segment-max key buffer (key 0 == -infinity sentinel).
// ---------------------------------------------------------------------------
__global__ void k_zinit(u32x4* __restrict__ p, int n4) {
    int i = blockIdx.x * blockDim.x + threadIdx.x;
    if (i < n4) {
        u32x4 z = {0u, 0u, 0u, 0u};
        p[i] = z;
    }
}

// ---------------------------------------------------------------------------
// Kernel 1: convert the three (tiny, L2-resident) weight matrices to bf16.
// ---------------------------------------------------------------------------
__global__ void k_cvt_weights(const float* __restrict__ W1,
                              const float* __restrict__ W2e,
                              const float* __restrict__ W2v,
                              unsigned short* __restrict__ W1b,
                              unsigned short* __restrict__ W2eb,
                              unsigned short* __restrict__ W2vb) {
    int i = blockIdx.x * blockDim.x + threadIdx.x;
    if (i < C_MID * C_IN)  W1b[i]  = f2bf(W1[i]);
    if (i < C_OUT * C_IN)  W2eb[i] = f2bf(W2e[i]);
    if (i < C_OUT * C_MID) W2vb[i] = f2bf(W2v[i]);
}

// ---------------------------------------------------------------------------
// Kernel 2: z = x @ W1^T + b1 fused with segment-max scatter (never
// materializes the 1 GB z tensor). Block = 512 thr (16 waves), tile 128x256,
// K=128. Wave w owns rows [16*(w/2), +16) and N-half (w&1)*128:
// 8 N-tiles -> acc = 64 VGPRs/wave (keeps allocation < 256, no VGPR-MSB).
// ---------------------------------------------------------------------------
__global__ __launch_bounds__(512) void k_gemm1_scatter(
    const float* __restrict__ x, const int* __restrict__ vid,
    const unsigned short* __restrict__ W1b, const float* __restrict__ b1,
    unsigned int* __restrict__ zkeys) {
    __shared__ unsigned short sW[C_MID * C_IN];  // 64 KB bf16 W1 [out][in]
    __shared__ float sB1[C_MID];

    const int tid = threadIdx.x;
    {
        const u32x4* src = (const u32x4*)W1b;
        u32x4* dst = (u32x4*)sW;
#pragma unroll
        for (int i = 0; i < (C_MID * C_IN / 8) / 512; i++)
            dst[tid + i * 512] = src[tid + i * 512];
        if (tid < C_MID) sB1[tid] = b1[tid];
    }
    __syncthreads();

    const int wave  = tid >> 5;
    const int lane  = tid & 31;
    const int h     = lane >> 4;
    const int ln    = lane & 15;
    const int nbase = (wave & 1) * 128;           // N half owned by this wave
    const int m0    = blockIdx.x * 128 + (wave >> 1) * 16;

    const v8f vzero = {0.f, 0.f, 0.f, 0.f, 0.f, 0.f, 0.f, 0.f};
    v8f acc[8];
#pragma unroll
    for (int t = 0; t < 8; t++) acc[t] = vzero;

    const float* rp = x + (size_t)(m0 + ln) * C_IN;
#pragma unroll
    for (int kk = 0; kk < 4; kk++) {
        const int k0 = kk * 32;
        v16bf a = a_frag_from_f32(rp, k0, h);
#pragma unroll
        for (int nt = 0; nt < 8; nt++) {
            v16bf b = b_frag_from_lds(sW, nbase + nt * 16 + ln, C_IN, k0, h);
            acc[nt] = wmma_bf16(a, b, acc[nt]);
        }
    }

    // Epilogue: D layout -> VGPR r holds M = r (lanes 0-15) / r+8 (16-31),
    // N = ln. Add bias, map to monotone key, atomic-max into vertex row.
    int vr[8];
#pragma unroll
    for (int r = 0; r < 8; r++) vr[r] = vid[m0 + r + 8 * h];

#pragma unroll
    for (int nt = 0; nt < 8; nt++) {
        const int n = nbase + nt * 16 + ln;
        const float bias = sB1[n];
#pragma unroll
        for (int r = 0; r < 8; r++) {
            const float v = acc[nt][r] + bias;
            atomicMax(&zkeys[(size_t)vr[r] * C_MID + n], fkey(v));
        }
    }
}

// ---------------------------------------------------------------------------
// Kernel 3: z_vertex = decode(zkeys) @ W2v^T. Tile 128x128, K=256.
// ---------------------------------------------------------------------------
__global__ __launch_bounds__(256) void k_vertex_gemm(
    const unsigned int* __restrict__ zkeys,
    const unsigned short* __restrict__ W2vb,
    float* __restrict__ zvert) {
    __shared__ unsigned short sW[C_OUT * C_MID];  // 64 KB

    const int tid = threadIdx.x;
    {
        const u32x4* src = (const u32x4*)W2vb;
        u32x4* dst = (u32x4*)sW;
#pragma unroll
        for (int i = 0; i < (C_OUT * C_MID / 8) / 256; i++)
            dst[tid + i * 256] = src[tid + i * 256];
    }
    __syncthreads();

    const int wave = tid >> 5;
    const int lane = tid & 31;
    const int h = lane >> 4;
    const int ln = lane & 15;
    const int m0 = blockIdx.x * 128 + wave * 16;

    int arow = m0 + ln;
    if (arow >= NVERT) arow = NVERT - 1;  // clamp; tail rows never stored
    const unsigned* rp = zkeys + (size_t)arow * C_MID;

    const v8f vzero = {0.f, 0.f, 0.f, 0.f, 0.f, 0.f, 0.f, 0.f};
    v8f acc[8];
#pragma unroll
    for (int t = 0; t < 8; t++) acc[t] = vzero;

#pragma unroll
    for (int kk = 0; kk < 8; kk++) {
        const int k0 = kk * 32;
        v16bf a = a_frag_from_keys(rp, k0, h);
#pragma unroll
        for (int nt = 0; nt < 8; nt++) {
            v16bf b = b_frag_from_lds(sW, nt * 16 + ln, C_MID, k0, h);
            acc[nt] = wmma_bf16(a, b, acc[nt]);
        }
    }

#pragma unroll
    for (int r = 0; r < 8; r++) {
        const int mrow = m0 + r + 8 * h;
        if (mrow < NVERT) {
            float* op = zvert + (size_t)mrow * C_OUT;
#pragma unroll
            for (int nt = 0; nt < 8; nt++) op[nt * 16 + ln] = acc[nt][r];
        }
    }
}

// ---------------------------------------------------------------------------
// Kernel 4: out = x @ W2e^T + z_vertex[vertex_id]. Tile 128x128, K=128.
// ---------------------------------------------------------------------------
__global__ __launch_bounds__(256) void k_edge_out(
    const float* __restrict__ x, const int* __restrict__ vid,
    const unsigned short* __restrict__ W2eb,
    const float* __restrict__ zvert,
    float* __restrict__ out) {
    __shared__ unsigned short sW[C_OUT * C_IN];  // 32 KB

    const int tid = threadIdx.x;
    {
        const u32x4* src = (const u32x4*)W2eb;
        u32x4* dst = (u32x4*)sW;
#pragma unroll
        for (int i = 0; i < (C_OUT * C_IN / 8) / 256; i++)
            dst[tid + i * 256] = src[tid + i * 256];
    }
    __syncthreads();

    const int wave = tid >> 5;
    const int lane = tid & 31;
    const int h = lane >> 4;
    const int ln = lane & 15;
    const int m0 = blockIdx.x * 128 + wave * 16;

    const v8f vzero = {0.f, 0.f, 0.f, 0.f, 0.f, 0.f, 0.f, 0.f};
    v8f acc[8];
#pragma unroll
    for (int t = 0; t < 8; t++) acc[t] = vzero;

    const float* rp = x + (size_t)(m0 + ln) * C_IN;
#pragma unroll
    for (int kk = 0; kk < 4; kk++) {
        const int k0 = kk * 32;
        v16bf a = a_frag_from_f32(rp, k0, h);
#pragma unroll
        for (int nt = 0; nt < 8; nt++) {
            v16bf b = b_frag_from_lds(sW, nt * 16 + ln, C_IN, k0, h);
            acc[nt] = wmma_bf16(a, b, acc[nt]);
        }
    }

#pragma unroll
    for (int r = 0; r < 8; r++) {
        const int mrow = m0 + r + 8 * h;
        const float* zp = zvert + (size_t)vid[mrow] * C_OUT;
        float* op = out + (size_t)mrow * C_OUT;
#pragma unroll
        for (int nt = 0; nt < 8; nt++) {
            const int n = nt * 16 + ln;
            op[n] = acc[nt][r] + zp[n];
        }
    }
}

// ---------------------------------------------------------------------------
extern "C" void kernel_launch(void* const* d_in, const int* in_sizes, int n_in,
                              void* d_out, int out_size, void* d_ws, size_t ws_size,
                              hipStream_t stream) {
    (void)in_sizes; (void)n_in; (void)out_size; (void)ws_size;

    const float* x   = (const float*)d_in[0];
    const int*   vid = (const int*)d_in[1];
    const float* W1  = (const float*)d_in[2];
    const float* b1  = (const float*)d_in[3];
    const float* W2e = (const float*)d_in[4];
    const float* W2v = (const float*)d_in[5];
    float* out = (float*)d_out;

    // Workspace layout (all 16B-aligned): ~77 MB total.
    char* ws = (char*)d_ws;
    unsigned int*   zkeys = (unsigned int*)ws;                                   // V*256 u32
    float*          zvert = (float*)(ws + (size_t)NVERT * C_MID * 4);            // V*128 f32
    unsigned short* W1b   = (unsigned short*)(ws + (size_t)NVERT * C_MID * 4
                                                 + (size_t)NVERT * C_OUT * 4);
    unsigned short* W2eb  = W1b + C_MID * C_IN;
    unsigned short* W2vb  = W2eb + C_OUT * C_IN;

    const int n4 = NVERT * C_MID / 4;
    k_zinit<<<(n4 + 255) / 256, 256, 0, stream>>>((u32x4*)zkeys, n4);
    k_cvt_weights<<<(C_MID * C_IN + 255) / 256, 256, 0, stream>>>(W1, W2e, W2v,
                                                                  W1b, W2eb, W2vb);
    k_gemm1_scatter<<<E_EDGES / 128, 512, 0, stream>>>(x, vid, W1b, b1, zkeys);
    k_vertex_gemm<<<(NVERT + 127) / 128, 256, 0, stream>>>(zkeys, W2vb, zvert);
    k_edge_out<<<E_EDGES / 128, 256, 0, stream>>>(x, vid, W2eb, zvert, out);
}
